// MultiHeadAttention_21741124452445
// MI455X (gfx1250) — matrix operand order repaired and checked by reading, not compiled, not run on hardware
//
#include <hip/hip_runtime.h>
#include <hip/hip_bf16.h>
#include <stdint.h>

// ---------------------------------------------------------------------------
// MultiHeadAttention for MI455X (gfx1250): f16 WMMA pipeline
//   proj (Q,K,V head projections) -> flash attention -> output GEMM + bias
// Data movement: async global->LDS (ASYNCcnt) for pure-copy staging,
// VGPR path for transposed/converted staging.
// ---------------------------------------------------------------------------

#define N_BATCH 2
#define SEQ_L   4096
#define EMBED_D 1024
#define N_HEADS 16
#define HEAD_D  64
// softmax scale = EMBED^-0.5
#define SM_SCALE 0.03125f

typedef __attribute__((ext_vector_type(16))) _Float16 v16h;
typedef __attribute__((ext_vector_type(8)))  float    v8f;

union Frag16 {
    v16h     v;
    uint32_t u[8];
    _Float16 h[16];
};

__device__ __forceinline__ v8f wmma_f16(const v16h& a, const v16h& b, const v8f& c) {
    return __builtin_amdgcn_wmma_f32_16x16x32_f16(
        /*neg_a=*/false, a, /*neg_b=*/false, b,
        /*c_mod=*/(short)0, c, /*reuse_a=*/false, /*reuse_b=*/false);
}

// A-fragment K-pair base index (this file's contiguous k order, A and B alike, wave32):
//  lanes 0-15 : dwords 0..7 -> K = 0..14
//  lanes16-31 : dwords 0..7 -> K = 16..30
__device__ __forceinline__ int a_k0(int p, bool hi) {
    return (hi ? 16 : 0) + 2 * p;
}

// Flat LDS pointer -> 32-bit LDS byte address (flat aperture keeps the LDS
// offset in addr[31:0]); 64-bit global address for GV-mode async copies.
__device__ __forceinline__ uint32_t lds_addr32(const void* p) {
    return (uint32_t)(uintptr_t)p;
}
__device__ __forceinline__ uint64_t gaddr64(const void* p) {
    return (uint64_t)(uintptr_t)p;
}
// Async 16B global->LDS copy (GV mode: 64-bit vgpr address, no SGPR base).
// INST_OFFSET is added to BOTH the LDS and global addresses per the ISA.
__device__ __forceinline__ void async_copy_b128(uint32_t lds, uint64_t gaddr) {
    asm volatile("global_load_async_to_lds_b128 %0, %1, off"
                 :: "v"(lds), "v"(gaddr) : "memory");
}
__device__ __forceinline__ void async_copy_b128_off16(uint32_t lds, uint64_t gaddr) {
    asm volatile("global_load_async_to_lds_b128 %0, %1, off offset:16"
                 :: "v"(lds), "v"(gaddr) : "memory");
}
__device__ __forceinline__ void wait_async0() {
    asm volatile("s_wait_asynccnt 0x0" ::: "memory");
}

// ---------------------------------------------------------------------------
// Kernel 1: per-head projection  Out[n][h][l][d] = f16( X[n,l,h*64:] @ W^T )
// One wave computes a 16(row) x 64(col) tile.
// ---------------------------------------------------------------------------
__global__ __launch_bounds__(256) void proj_kernel(
    const float* __restrict__ X, const float* __restrict__ W,
    _Float16* __restrict__ Out)
{
    __shared__ _Float16 Wl[64][64];      // W as-is: Wl[e][d]
    __shared__ _Float16 Xs[8][16][64];   // per-wave X tile (f16)

    const int t = threadIdx.x;
    // ---- stage W (64x64 f32 -> f16) ----
    {
        const int row = t >> 2;            // 0..63
        const int cb  = (t & 3) * 16;      // 0,16,32,48
        const float4* wp = (const float4*)(W + row * 64 + cb);
        #pragma unroll
        for (int i = 0; i < 4; ++i) {
            float4 w4 = wp[i];
            Wl[row][cb + 4 * i + 0] = (_Float16)w4.x;
            Wl[row][cb + 4 * i + 1] = (_Float16)w4.y;
            Wl[row][cb + 4 * i + 2] = (_Float16)w4.z;
            Wl[row][cb + 4 * i + 3] = (_Float16)w4.w;
        }
    }

    const int wave = t >> 5, lane = t & 31;
    const int gid  = blockIdx.x * 8 + wave;        // 0 .. N*H*L/16-1
    const int lt   = gid & 255;                    // L/16 = 256 tiles
    const int h    = (gid >> 8) & 15;
    const int n    = gid >> 12;
    const int l0   = lt * 16;

    // ---- stage X tile (16 rows x 64 cols) ----
    #pragma unroll
    for (int i = 0; i < 8; ++i) {
        const int fi  = lane + 32 * i;             // 0..255
        const int row = fi >> 4;
        const int c4  = (fi & 15) * 4;
        const float4 x4 = *(const float4*)(
            X + (size_t)(n * SEQ_L + l0 + row) * EMBED_D + h * HEAD_D + c4);
        Xs[wave][row][c4 + 0] = (_Float16)x4.x;
        Xs[wave][row][c4 + 1] = (_Float16)x4.y;
        Xs[wave][row][c4 + 2] = (_Float16)x4.z;
        Xs[wave][row][c4 + 3] = (_Float16)x4.w;
    }
    __syncthreads();

    const int  mrow = lane & 15;
    const bool hi   = lane >= 16;

    // ---- A fragments: rows of X tile, K = d (0..31 and 32..63) ----
    Frag16 a0, a1;
    #pragma unroll
    for (int p = 0; p < 8; ++p) {
        const int k0 = a_k0(p, hi);
        a0.u[p] = *(const uint32_t*)&Xs[wave][mrow][k0];
        a1.u[p] = *(const uint32_t*)&Xs[wave][mrow][k0 + 32];
    }

    #pragma unroll
    for (int c = 0; c < 4; ++c) {
        // B[kk][col] = W[16c+col][kk]  (x @ W^T)
        Frag16 b0, b1;
        const uint32_t* wrow = (const uint32_t*)&Wl[16 * c + mrow][0];
        #pragma unroll
        for (int v = 0; v < 8; ++v) {
            const int kk0 = (hi ? 16 : 0) + 2 * v;
            b0.u[v] = wrow[kk0 >> 1];
            b1.u[v] = wrow[(kk0 + 32) >> 1];
        }
        v8f acc = {0.f, 0.f, 0.f, 0.f, 0.f, 0.f, 0.f, 0.f};
        acc = wmma_f16(a0.v, b0.v, acc);
        acc = wmma_f16(a1.v, b1.v, acc);
        #pragma unroll
        for (int r = 0; r < 8; ++r) {
            const int row = r + (hi ? 8 : 0);
            Out[((size_t)(n * N_HEADS + h) * SEQ_L + l0 + row) * HEAD_D
                + 16 * c + mrow] = (_Float16)acc[r];
        }
    }
}

// ---------------------------------------------------------------------------
// Kernel 2: flash attention per (n, h). Block = 8 waves = 128 query rows.
// K tile async-copied global->LDS (ASYNCcnt); V staged transposed through
// VGPRs; online softmax; P routed via LDS (C-layout -> A-layout) for P·V.
// ---------------------------------------------------------------------------
__global__ __launch_bounds__(256) void attn_kernel(
    const _Float16* __restrict__ Qp, const _Float16* __restrict__ Kp,
    const _Float16* __restrict__ Vp, _Float16* __restrict__ AttOut)
{
    __shared__ _Float16 Kt[32][64];       // K tile, row-major in d
    __shared__ _Float16 Vt[64][32];       // V tile, transposed: Vt[d][kk]
    __shared__ _Float16 Pl[8][16][32];    // per-wave P staging

    const int t = threadIdx.x, wave = t >> 5, lane = t & 31;
    const int n = blockIdx.z, h = blockIdx.y;
    const int q0 = blockIdx.x * 128 + wave * 16;
    const size_t headBase = (size_t)(n * N_HEADS + h) * SEQ_L * HEAD_D;
    const int  mrow = lane & 15;
    const bool hi   = lane >= 16;

    // ---- Q A-fragments (persist across the whole K loop) ----
    Frag16 aq0, aq1;
    {
        const _Float16* qrow = Qp + headBase + (size_t)(q0 + mrow) * HEAD_D;
        #pragma unroll
        for (int p = 0; p < 8; ++p) {
            const int k0 = a_k0(p, hi);
            aq0.u[p] = *(const uint32_t*)(qrow + k0);
            aq1.u[p] = *(const uint32_t*)(qrow + k0 + 32);
        }
    }

    float mrun[8], lrun[8];
    v8f acc[4];
    #pragma unroll
    for (int r = 0; r < 8; ++r) { mrun[r] = -3.0e38f; lrun[r] = 0.f; }
    #pragma unroll
    for (int c = 0; c < 4; ++c)
        acc[c] = (v8f){0.f, 0.f, 0.f, 0.f, 0.f, 0.f, 0.f, 0.f};

    for (int kb = 0; kb < SEQ_L; kb += 32) {
        __syncthreads();   // prior iteration finished reading Kt/Vt
        // ---- stage K (async copy) and V (transposed via VGPRs) tiles ----
        {
            const int row = t >> 3;          // 0..31
            const int dp  = (t & 7) * 8;     // 0..56
            // K: pure 16B copy -> async global->LDS, no VGPR data
            async_copy_b128(lds_addr32(&Kt[row][dp]),
                            gaddr64(Kp + headBase + (size_t)(kb + row) * HEAD_D + dp));
            // V: needs transpose -> VGPR path
            const uint4 vv = *(const uint4*)(
                Vp + headBase + (size_t)(kb + row) * HEAD_D + dp);
            const _Float16* vh = (const _Float16*)&vv;
            #pragma unroll
            for (int i = 0; i < 8; ++i) Vt[dp + i][row] = vh[i];
            if (kb + 32 < SEQ_L) {
                __builtin_prefetch(Vp + headBase + (size_t)(kb + 32 + row) * HEAD_D + dp, 0, 1);
            }
        }
        wait_async0();     // K tile landed in LDS for this wave's requests
        __syncthreads();   // all waves' K/V staging visible block-wide

        // ---- S = (Q K^T) * scale : two 16x16 tiles ----
        v8f s[2];
        #pragma unroll
        for (int ck = 0; ck < 2; ++ck) {
            Frag16 bk0, bk1;     // B[dd][col] = K[16ck+col][dd]
            const uint32_t* kr = (const uint32_t*)&Kt[16 * ck + mrow][0];
            #pragma unroll
            for (int v = 0; v < 8; ++v) {
                const int kk0 = (hi ? 16 : 0) + 2 * v;
                bk0.u[v] = kr[kk0 >> 1];
                bk1.u[v] = kr[(kk0 + 32) >> 1];
            }
            v8f sc = {0.f, 0.f, 0.f, 0.f, 0.f, 0.f, 0.f, 0.f};
            sc = wmma_f16(aq0.v, bk0.v, sc);
            sc = wmma_f16(aq1.v, bk1.v, sc);
            s[ck] = sc * SM_SCALE;
        }

        // ---- online softmax (each 16-lane half owns rows r(+8) ) ----
        float mnew[8], alpha[8], psum[8];
        #pragma unroll
        for (int r = 0; r < 8; ++r) {
            float mx = fmaxf(s[0][r], s[1][r]);
            #pragma unroll
            for (int off = 1; off < 16; off <<= 1)
                mx = fmaxf(mx, __shfl_xor(mx, off, 32));
            mnew[r]  = fmaxf(mrun[r], mx);
            alpha[r] = __expf(mrun[r] - mnew[r]);
        }
        #pragma unroll
        for (int r = 0; r < 8; ++r) {
            const float p0 = __expf(s[0][r] - mnew[r]);
            const float p1 = __expf(s[1][r] - mnew[r]);
            const int row = r + (hi ? 8 : 0);
            Pl[wave][row][mrow]      = (_Float16)p0;
            Pl[wave][row][16 + mrow] = (_Float16)p1;
            float ps = p0 + p1;
            #pragma unroll
            for (int off = 1; off < 16; off <<= 1)
                ps += __shfl_xor(ps, off, 32);
            psum[r] = ps;
        }
        #pragma unroll
        for (int r = 0; r < 8; ++r) {
            lrun[r] = lrun[r] * alpha[r] + psum[r];
            mrun[r] = mnew[r];
        }
        #pragma unroll
        for (int c = 0; c < 4; ++c)
            #pragma unroll
            for (int r = 0; r < 8; ++r) acc[c][r] *= alpha[r];

        // per-wave LDS ops are in-order; fence against compiler reordering
        asm volatile("s_wait_dscnt 0x0" ::: "memory");

        // ---- P as A-fragment, then O += P V ----
        Frag16 ap;
        {
            const uint32_t* pr = (const uint32_t*)&Pl[wave][mrow][0];
            #pragma unroll
            for (int p = 0; p < 8; ++p) ap.u[p] = pr[a_k0(p, hi) >> 1];
        }
        #pragma unroll
        for (int c = 0; c < 4; ++c) {
            Frag16 bv;   // B[kk][col] = V[kk][16c+col] -> Vt[16c+col][kk]
            const uint32_t* vr = (const uint32_t*)&Vt[16 * c + mrow][0];
            #pragma unroll
            for (int v = 0; v < 8; ++v) bv.u[v] = vr[(hi ? 8 : 0) + v];
            acc[c] = wmma_f16(ap.v, bv.v, acc[c]);
        }
    }

    // ---- epilogue: normalize and store [n][l][h*64+d] as f16 ----
    #pragma unroll
    for (int c = 0; c < 4; ++c) {
        #pragma unroll
        for (int r = 0; r < 8; ++r) {
            const int row = r + (hi ? 8 : 0);
            const float o = acc[c][r] / lrun[r];
            AttOut[(size_t)(n * SEQ_L + q0 + row) * EMBED_D
                   + h * HEAD_D + 16 * c + mrow] = (_Float16)o;
        }
    }
}

// ---------------------------------------------------------------------------
// Kernel 3: Out[8192,1024] = Att(f16) @ Wo^T + bo   (fp32 out)
// Block tile 128x64; A tile async-copied global->LDS, Wo converted f32->f16
// through VGPRs; 4 WMMA per wave per 32-deep K chunk.
// ---------------------------------------------------------------------------
__global__ __launch_bounds__(256) void outproj_kernel(
    const _Float16* __restrict__ A, const float* __restrict__ Wo,
    const float* __restrict__ bo, float* __restrict__ Out)
{
    __shared__ _Float16 As[128][32];
    __shared__ _Float16 Bs[64][32];   // Bs[col][kk] = f16(Wo[col0+col][kb+kk])

    const int t = threadIdx.x, wave = t >> 5, lane = t & 31;
    const int row0 = blockIdx.x * 128, col0 = blockIdx.y * 64;
    const int  mrow = lane & 15;
    const bool hi   = lane >= 16;

    v8f acc[4];
    #pragma unroll
    for (int c = 0; c < 4; ++c)
        acc[c] = (v8f){0.f, 0.f, 0.f, 0.f, 0.f, 0.f, 0.f, 0.f};

    for (int kb = 0; kb < EMBED_D; kb += 32) {
        __syncthreads();
        {   // stage A tile: 128 rows x 32 halves, two async 16B copies/thread
            const int row  = t >> 1;
            const int part = (t & 1) * 16;
            const uint32_t ldsA = lds_addr32(&As[row][part]);
            const uint64_t gA   = gaddr64(A + (size_t)(row0 + row) * EMBED_D + kb + part);
            async_copy_b128(ldsA, gA);          // halves [part, part+8)
            async_copy_b128_off16(ldsA, gA);    // halves [part+8, part+16)
            // stage B: 64 cols x 32 kk (f32 -> f16, VGPR path)
            const int col = t >> 2;
            const int kp  = (t & 3) * 8;
            const float4* wp = (const float4*)(
                Wo + (size_t)(col0 + col) * EMBED_D + kb + kp);
            const float4 w0 = wp[0], w1 = wp[1];
            Bs[col][kp + 0] = (_Float16)w0.x;
            Bs[col][kp + 1] = (_Float16)w0.y;
            Bs[col][kp + 2] = (_Float16)w0.z;
            Bs[col][kp + 3] = (_Float16)w0.w;
            Bs[col][kp + 4] = (_Float16)w1.x;
            Bs[col][kp + 5] = (_Float16)w1.y;
            Bs[col][kp + 6] = (_Float16)w1.z;
            Bs[col][kp + 7] = (_Float16)w1.w;
        }
        wait_async0();
        __syncthreads();

        Frag16 a;
        const uint32_t* ar = (const uint32_t*)&As[wave * 16 + mrow][0];
        #pragma unroll
        for (int p = 0; p < 8; ++p) a.u[p] = ar[a_k0(p, hi) >> 1];

        #pragma unroll
        for (int c = 0; c < 4; ++c) {
            Frag16 b;
            const uint32_t* br = (const uint32_t*)&Bs[16 * c + mrow][0];
            #pragma unroll
            for (int v = 0; v < 8; ++v) b.u[v] = br[(hi ? 8 : 0) + v];
            acc[c] = wmma_f16(a.v, b.v, acc[c]);
        }
    }

    #pragma unroll
    for (int c = 0; c < 4; ++c) {
        const int col  = col0 + 16 * c + mrow;
        const float bias = bo[col];
        #pragma unroll
        for (int r = 0; r < 8; ++r) {
            const int row = row0 + wave * 16 + r + (hi ? 8 : 0);
            Out[(size_t)row * EMBED_D + col] = acc[c][r] + bias;
        }
    }
}

// ---------------------------------------------------------------------------
extern "C" void kernel_launch(void* const* d_in, const int* in_sizes, int n_in,
                              void* d_out, int out_size, void* d_ws, size_t ws_size,
                              hipStream_t stream) {
    (void)in_sizes; (void)n_in; (void)out_size; (void)ws_size;

    const float* value = (const float*)d_in[0];
    const float* key   = (const float*)d_in[1];
    const float* query = (const float*)d_in[2];
    const float* Wv    = (const float*)d_in[3];
    const float* Wk    = (const float*)d_in[4];
    const float* Wq    = (const float*)d_in[5];
    const float* Wo    = (const float*)d_in[6];
    const float* bo    = (const float*)d_in[7];
    float*       out   = (float*)d_out;

    const size_t elems = (size_t)N_BATCH * N_HEADS * SEQ_L * HEAD_D; // 8,388,608
    _Float16* Qp  = (_Float16*)d_ws;
    _Float16* Kp  = Qp + elems;
    _Float16* Vp  = Kp + elems;
    _Float16* Att = Vp + elems;   // N*L*EMBED == elems

    const int projBlocks = (int)((size_t)N_BATCH * N_HEADS * SEQ_L / 16 / 8); // 1024
    proj_kernel<<<projBlocks, 256, 0, stream>>>(query, Wq, Qp);
    proj_kernel<<<projBlocks, 256, 0, stream>>>(key,   Wk, Kp);
    proj_kernel<<<projBlocks, 256, 0, stream>>>(value, Wv, Vp);

    dim3 ga(SEQ_L / 128, N_HEADS, N_BATCH);
    attn_kernel<<<ga, 256, 0, stream>>>(Qp, Kp, Vp, Att);

    dim3 gg(N_BATCH * SEQ_L / 128, EMBED_D / 64);
    outproj_kernel<<<gg, 256, 0, stream>>>(Att, Wo, bo, out);
}
